// FastRCNNPredictor_3788161155092
// MI455X (gfx1250) — compile-verified, matrix operand order checked by reading, not run on hardware
//
#include <hip/hip_runtime.h>
#include <hip/hip_bf16.h>

typedef __attribute__((ext_vector_type(16))) _Float16 v16h;
typedef __attribute__((ext_vector_type(8)))  _Float16 v8h;
typedef __attribute__((ext_vector_type(8)))  float    v8f;

#define NROI 512
#define CTOT 2048
#define GRP  4
#define CG   512
#define II   256
#define SS   49
#define SP   64
#define NCLS 81
#define NBOX 324

// LDS row pitches (in halves for f16 buffers, floats for attn_f)
#define XP  520   // x tiles: [64 s][512 c + pad]
#define QP  264   // q_t/k_t/y_t: [64 s][256 i + pad]
#define VP  72    // v: [256 i][64 t + pad]
#define AFP 68    // attn logits f32: [64 s][64 t + pad]
#define AHP 72    // attn probs f16:  [64 s][64 t + pad]

// LDS byte offsets (all 16B aligned)
#define OFF_XT   0u
#define OFF_X2T  66560u
#define OFF_QT   133120u
#define OFF_KT   166912u
#define OFF_VM   200704u
#define OFF_YT   237568u
#define OFF_AF   271360u
#define OFF_AH   288768u
#define LDS_TOTAL 297984u

__device__ __forceinline__ v16h frag16(const _Float16* p) {
  // CDNA5 16-bit A/B fragment: lanes 0-15 hold K=[k0..k0+7],[k0+16..k0+23],
  // lanes 16-31 hold K=[k0+8..k0+15],[k0+24..k0+31]; caller pre-adds hi*8.
  v8h lo = *(const v8h*)(p);
  v8h hv = *(const v8h*)(p + 16);
  v16h r;
#pragma unroll
  for (int i = 0; i < 8; ++i) { r[i] = lo[i]; r[i + 8] = hv[i]; }
  return r;
}

__device__ __forceinline__ v8f wmma16(v16h a, v16h b, v8f c) {
  return __builtin_amdgcn_wmma_f32_16x16x32_f16(false, a, false, b, (short)0, c,
                                                false, false);
}

// One grouped non-local block on a [512 c x 64 s] LDS-resident tile.
// in/out layout: [s][c] (c contiguous, pitch XP). Never in-place.
__device__ __forceinline__ void nl_block(
    const _Float16* __restrict__ Wqh, const _Float16* __restrict__ Wkh,
    const _Float16* __restrict__ Wvh, const _Float16* __restrict__ Woh,
    const float* __restrict__ bq, const float* __restrict__ bk,
    const float* __restrict__ bv, const float* __restrict__ bo,
    _Float16* in, _Float16* out,
    _Float16* q_t, _Float16* k_t, _Float16* v_m,
    float* attn_f, _Float16* attn_h, _Float16* y_t,
    int wave, int lane, int tid)
{
  const int r  = lane & 15;
  const int hi = lane >> 4;

  // ---- GEMM1: q/k/v[256 x 64] = W[256x512] * x ----
  // Wave = one i-tile (mt = wave); q, k, v computed together so the 4 shared
  // B (x_t) fragments are loaded once per k-step: 14 b128 loads / 12 WMMAs.
  {
    const int mt = wave;                 // 0..15
    v8f accq[4] = {{}, {}, {}, {}};
    v8f acck[4] = {{}, {}, {}, {}};
    v8f accv[4] = {{}, {}, {}, {}};
    const _Float16* qrow = Wqh + (size_t)(mt * 16 + r) * CG + hi * 8;
    const _Float16* krow = Wkh + (size_t)(mt * 16 + r) * CG + hi * 8;
    const _Float16* vrow = Wvh + (size_t)(mt * 16 + r) * CG + hi * 8;
    const _Float16* brow = in + r * XP + hi * 8;
    for (int k0 = 0; k0 < CG; k0 += 32) {
      const v16h aq = frag16(qrow + k0);
      const v16h ak = frag16(krow + k0);
      const v16h av = frag16(vrow + k0);
#pragma unroll
      for (int nt = 0; nt < 4; ++nt) {
        const v16h b = frag16(brow + nt * 16 * XP + k0);
        accq[nt] = wmma16(aq, b, accq[nt]);
        acck[nt] = wmma16(ak, b, acck[nt]);
        accv[nt] = wmma16(av, b, accv[nt]);
      }
    }
    const int i0 = mt * 16 + hi * 8;
#pragma unroll
    for (int nt = 0; nt < 4; ++nt) {
      // q, k stored transposed: q_t/k_t[s][i] (8 contiguous halves per lane)
      v8h hq, hk;
#pragma unroll
      for (int e = 0; e < 8; ++e) {
        hq[e] = (_Float16)(accq[nt][e] + bq[i0 + e]);
        hk[e] = (_Float16)(acck[nt][e] + bk[i0 + e]);
      }
      *(v8h*)(q_t + (nt * 16 + r) * QP + i0) = hq;
      *(v8h*)(k_t + (nt * 16 + r) * QP + i0) = hk;
      // v natural: v_m[i][t]
      const int t = nt * 16 + r;
#pragma unroll
      for (int e = 0; e < 8; ++e)
        v_m[(i0 + e) * VP + t] = (_Float16)(accv[nt][e] + bv[i0 + e]);
    }
  }
  __syncthreads();

  // ---- GEMM2: logits[s][t] = q^T k   (A=q_t rows, B=k_t rows, K=256) ----
  for (int tt = wave; tt < 16; tt += 16) {
    const int st = tt >> 2, t2 = tt & 3;
    v8f acc = {};
    const _Float16* arow = q_t + (st * 16 + r) * QP + hi * 8;
    const _Float16* brow = k_t + (t2 * 16 + r) * QP + hi * 8;
#pragma unroll
    for (int k0 = 0; k0 < II; k0 += 32)
      acc = wmma16(frag16(arow + k0), frag16(brow + k0), acc);
    float* col = attn_f + (st * 16 + hi * 8) * AFP + t2 * 16 + r;
#pragma unroll
    for (int e = 0; e < 8; ++e) col[e * AFP] = acc[e];
  }
  __syncthreads();

  // ---- softmax over t: 8 lanes per row, all 512 threads active ----
  {
    const int row = tid >> 3;            // 0..63
    const int sub = tid & 7;
    float*    rp   = attn_f + row * AFP;
    _Float16* hrow = attn_h + row * AHP;
    if (row < SS) {
      float m = -1e30f;
      for (int t = sub; t < SS; t += 8) m = fmaxf(m, rp[t]);
#pragma unroll
      for (int o = 1; o < 8; o <<= 1) m = fmaxf(m, __shfl_xor(m, o, 8));
      float s = 0.f;
      for (int t = sub; t < SS; t += 8) s += __expf(rp[t] - m);
#pragma unroll
      for (int o = 1; o < 8; o <<= 1) s += __shfl_xor(s, o, 8);
      const float inv = 1.f / s;
      for (int t = sub; t < SP; t += 8)
        hrow[t] = (_Float16)((t < SS) ? __expf(rp[t] - m) * inv : 0.f);
    } else {
      for (int t = sub; t < SP; t += 8) hrow[t] = (_Float16)0.f;
    }
  }
  __syncthreads();

  // ---- GEMM3: y_t[s][i] = attn[s][t] * v[i][t] (K=64) ----
  // One i-tile per wave; B (v) fragment reused across the 4 s-tiles.
  for (int it = wave; it < 16; it += 16) {
    v8f acc[4] = {{}, {}, {}, {}};
    const _Float16* brow = v_m + (it * 16 + r) * VP + hi * 8;
    const _Float16* arow = attn_h + r * AHP + hi * 8;
#pragma unroll
    for (int k0 = 0; k0 < SP; k0 += 32) {
      const v16h b = frag16(brow + k0);
#pragma unroll
      for (int st = 0; st < 4; ++st)
        acc[st] = wmma16(frag16(arow + st * 16 * AHP + k0), b, acc[st]);
    }
#pragma unroll
    for (int st = 0; st < 4; ++st) {
      _Float16* col = y_t + (st * 16 + hi * 8) * QP + it * 16 + r;
#pragma unroll
      for (int e = 0; e < 8; ++e) col[e * QP] = (_Float16)acc[st][e];
    }
  }
  __syncthreads();

  // ---- GEMM4: z[c][s] = Wo[512x256] * y + bias + residual, ReLU ----
  // Wave = two c-tiles (wave, wave+16); the 4 B (y_t) fragments are shared:
  // 12 b128 loads / 8 WMMAs per k-step.
  {
    v8f acc0[4] = {{}, {}, {}, {}};      // ct = wave
    v8f acc1[4] = {{}, {}, {}, {}};      // ct = wave + 16
    const _Float16* arow0 = Woh + (size_t)(wave * 16 + r) * II + hi * 8;
    const _Float16* arow1 = arow0 + (size_t)16 * 16 * II;
    const _Float16* brow  = y_t + r * QP + hi * 8;
    for (int k0 = 0; k0 < II; k0 += 32) {
      const v16h a0 = frag16(arow0 + k0);
      const v16h a1 = frag16(arow1 + k0);
#pragma unroll
      for (int st = 0; st < 4; ++st) {
        const v16h b = frag16(brow + st * 16 * QP + k0);
        acc0[st] = wmma16(a0, b, acc0[st]);
        acc1[st] = wmma16(a1, b, acc1[st]);
      }
    }
#pragma unroll
    for (int half = 0; half < 2; ++half) {
      const int c0 = (wave + half * 16) * 16 + hi * 8;
      v8f* accs = half ? acc1 : acc0;
#pragma unroll
      for (int st = 0; st < 4; ++st) {
        const int s = st * 16 + r;
        _Float16* dst = out + s * XP + c0;
        v8h hv;
        if (s < SS) {
          const _Float16* res = in + s * XP + c0;
#pragma unroll
          for (int e = 0; e < 8; ++e) {
            float z = accs[st][e] + bo[c0 + e] + (float)res[e];
            hv[e] = (_Float16)fmaxf(z, 0.f);
          }
        } else {
#pragma unroll
          for (int e = 0; e < 8; ++e) hv[e] = (_Float16)0.f;
        }
        *(v8h*)dst = hv;
      }
    }
  }
  __syncthreads();
}

// ---------------------------------------------------------------------------
__global__ void cvt_f32_to_f16_kernel(const float* __restrict__ src,
                                      _Float16* __restrict__ dst, int n) {
  int i = blockIdx.x * 256 + threadIdx.x;
  if (i < n) dst[i] = (_Float16)src[i];
}

// ---------------------------------------------------------------------------
// One workgroup per (n, g): full 3-block non-local chain + pooling.
__global__ __launch_bounds__(512) void frcnn_nonlocal_kernel(
    const float* __restrict__ x,
    const _Float16* __restrict__ wq_h, const _Float16* __restrict__ wk_h,
    const _Float16* __restrict__ wv_h, const _Float16* __restrict__ wo_h,
    const float* __restrict__ bq, const float* __restrict__ bk,
    const float* __restrict__ bv, const float* __restrict__ bo,
    float* __restrict__ pc, float* __restrict__ pr)
{
  extern __shared__ char smem[];
  _Float16* x_t    = (_Float16*)(smem + OFF_XT);
  _Float16* x2_t   = (_Float16*)(smem + OFF_X2T);
  _Float16* q_t    = (_Float16*)(smem + OFF_QT);
  _Float16* k_t    = (_Float16*)(smem + OFF_KT);
  _Float16* v_m    = (_Float16*)(smem + OFF_VM);
  _Float16* y_t    = (_Float16*)(smem + OFF_YT);
  float*    attn_f = (float*)   (smem + OFF_AF);
  _Float16* attn_h = (_Float16*)(smem + OFF_AH);

  const int n    = blockIdx.x >> 2;
  const int g    = blockIdx.x & 3;
  const int tid  = threadIdx.x;
  const int wave = tid >> 5;
  const int lane = tid & 31;

  // Load x tile (transpose [c][s] -> x_t[s][c], fp32 -> f16), zero-pad s>=49.
  const float* xg = x + ((size_t)n * CTOT + (size_t)g * CG) * SS;
  for (int e = tid; e < CG * SS; e += 512) {
    const int c = e / SS;
    const int s = e - c * SS;
    x_t[s * XP + c] = (_Float16)xg[e];
  }
  for (int e = tid; e < (SP - SS) * CG; e += 512) {
    const int s = SS + (e >> 9);
    x_t[s * XP + (e & 511)] = (_Float16)0.f;
  }
  __syncthreads();

  // block 0 (shared): x_t -> x2_t
  nl_block(wq_h, wk_h, wv_h, wo_h, bq, bk, bv, bo,
           x_t, x2_t, q_t, k_t, v_m, attn_f, attn_h, y_t, wave, lane, tid);
  // block 1 (cls): x2_t -> x_t
  nl_block(wq_h + 1 * II * CG, wk_h + 1 * II * CG, wv_h + 1 * II * CG,
           wo_h + 1 * CG * II, bq + II, bk + II, bv + II, bo + CG,
           x2_t, x_t, q_t, k_t, v_m, attn_f, attn_h, y_t, wave, lane, tid);
  {  // pool cls branch: mean over 49 positions for this group's channels
    const int c = tid;
    float s = 0.f;
    for (int t = 0; t < SS; ++t) s += (float)x_t[t * XP + c];
    pc[(size_t)n * CTOT + (size_t)g * CG + c] = s * (1.f / (float)SS);
  }
  __syncthreads();
  // block 2 (reg): x2_t -> x_t
  nl_block(wq_h + 2 * II * CG, wk_h + 2 * II * CG, wv_h + 2 * II * CG,
           wo_h + 2 * CG * II, bq + 2 * II, bk + 2 * II, bv + 2 * II,
           bo + 2 * CG,
           x2_t, x_t, q_t, k_t, v_m, attn_f, attn_h, y_t, wave, lane, tid);
  {  // pool reg branch
    const int c = tid;
    float s = 0.f;
    for (int t = 0; t < SS; ++t) s += (float)x_t[t * XP + c];
    pr[(size_t)n * CTOT + (size_t)g * CG + c] = s * (1.f / (float)SS);
  }
}

// ---------------------------------------------------------------------------
// Final linears: one wave per 16x16 output tile, WMMA over K=2048.
#define CLS_TILES 6   // ceil(81/16)
#define REG_TILES 21  // ceil(324/16)

__global__ __launch_bounds__(32) void frcnn_head_kernel(
    const float* __restrict__ pc, const float* __restrict__ pr,
    const float* __restrict__ Wc, const float* __restrict__ bc,
    const float* __restrict__ Wb, const float* __restrict__ bb,
    float* __restrict__ out)
{
  const int mt = blockIdx.x;   // n-tile 0..31
  const int yt = blockIdx.y;   // 0..26
  const float* P; const float* W; const float* bias; float* O;
  int ncols, nt;
  if (yt < CLS_TILES) { P = pc; W = Wc; bias = bc; O = out;
                        ncols = NCLS; nt = yt; }
  else                { P = pr; W = Wb; bias = bb; O = out + NROI * NCLS;
                        ncols = NBOX; nt = yt - CLS_TILES; }

  const int lane = threadIdx.x;
  const int r  = lane & 15;
  const int hi = lane >> 4;
  const int mrow = mt * 16 + r;        // roi row (<512 always)
  const int ncol = nt * 16 + r;        // output column (guarded)
  const float bsel = (ncol < ncols) ? 1.f : 0.f;
  const float* arow = P + (size_t)mrow * CTOT + hi * 8;
  const float* brow = W + (size_t)((ncol < ncols) ? ncol : 0) * CTOT + hi * 8;

  v8f acc = {};
  for (int k0 = 0; k0 < CTOT; k0 += 32) {
    v16h a, b;
#pragma unroll
    for (int e = 0; e < 8; ++e) {
      a[e]     = (_Float16)arow[k0 + e];
      a[e + 8] = (_Float16)arow[k0 + 16 + e];
      b[e]     = (_Float16)(brow[k0 + e] * bsel);
      b[e + 8] = (_Float16)(brow[k0 + 16 + e] * bsel);
    }
    acc = wmma16(a, b, acc);
  }
  if (ncol < ncols) {
#pragma unroll
    for (int e = 0; e < 8; ++e) {
      const int nr = mt * 16 + hi * 8 + e;
      O[(size_t)nr * ncols + ncol] = acc[e] + bias[ncol];
    }
  }
}

// ---------------------------------------------------------------------------
extern "C" void kernel_launch(void* const* d_in, const int* in_sizes, int n_in,
                              void* d_out, int out_size, void* d_ws, size_t ws_size,
                              hipStream_t stream) {
  (void)in_sizes; (void)n_in; (void)out_size; (void)ws_size;
  const float* x  = (const float*)d_in[0];
  const float* Wq = (const float*)d_in[1];
  const float* bq = (const float*)d_in[2];
  const float* Wk = (const float*)d_in[3];
  const float* bk = (const float*)d_in[4];
  const float* Wv = (const float*)d_in[5];
  const float* bv = (const float*)d_in[6];
  const float* Wo = (const float*)d_in[7];
  const float* bo = (const float*)d_in[8];
  const float* Wc = (const float*)d_in[9];
  const float* bc = (const float*)d_in[10];
  const float* Wb = (const float*)d_in[11];
  const float* bb = (const float*)d_in[12];
  float* out = (float*)d_out;

  // workspace layout
  const size_t WN = (size_t)3 * II * CG;  // per qkv-stack element count
  char* ws = (char*)d_ws;
  _Float16* wq_h = (_Float16*)(ws);                    // 3*256*512 f16
  _Float16* wk_h = (_Float16*)(ws + WN * 2);
  _Float16* wv_h = (_Float16*)(ws + WN * 4);
  _Float16* wo_h = (_Float16*)(ws + WN * 6);           // 3*512*256 f16
  float*    pc   = (float*)   (ws + WN * 8);           // 512*2048 f32
  float*    pr   = (float*)   (ws + WN * 8 + (size_t)NROI * CTOT * 4);

  // 1) convert weights to f16 (L2-resident, shared by all workgroups)
  const int wn = (int)WN;
  cvt_f32_to_f16_kernel<<<(wn + 255) / 256, 256, 0, stream>>>(Wq, wq_h, wn);
  cvt_f32_to_f16_kernel<<<(wn + 255) / 256, 256, 0, stream>>>(Wk, wk_h, wn);
  cvt_f32_to_f16_kernel<<<(wn + 255) / 256, 256, 0, stream>>>(Wv, wv_h, wn);
  cvt_f32_to_f16_kernel<<<(wn + 255) / 256, 256, 0, stream>>>(Wo, wo_h, wn);

  // 2) fused 3x non-local chain + pooling: one WG per (n, g)
  frcnn_nonlocal_kernel<<<NROI * GRP, 512, LDS_TOTAL, stream>>>(
      x, wq_h, wk_h, wv_h, wo_h, bq, bk, bv, bo, pc, pr);

  // 3) classification / bbox heads
  dim3 hgrid(NROI / 16, CLS_TILES + REG_TILES);
  frcnn_head_kernel<<<hgrid, 32, 0, stream>>>(pc, pr, Wc, bc, Wb, bb, out);
}
